// EdgeModel_24756191494617
// MI455X (gfx1250) — matrix operand order repaired and checked by reading, not compiled, not run on hardware
//
#include <hip/hip_runtime.h>
#include <hip/hip_bf16.h>

typedef __attribute__((ext_vector_type(16))) __bf16 v16bf;
typedef __attribute__((ext_vector_type(8)))  __bf16 bf16x8;
typedef __attribute__((ext_vector_type(4)))  __bf16 bf16x4;
typedef __attribute__((ext_vector_type(8)))  float  v8f;

#define F       128   // feature width (F_E = F_U = ... = 128)
#define KTOT    512   // concat width
#define ROWS    32    // edges per workgroup
#define HPITCH  520   // staged row pitch in bf16 elems (512 + 8 pad)
#define H2PITCH 136   // intermediate row pitch in bf16 elems (128 + 8 pad)

// ---------------------------------------------------------------------------
// Pack fp32 row-major weight [K, 128] into bf16 WMMA B-operand tiles.
// Layout: for (kt, wave w, lane, j): element = W[kt*32 + (j>>3)*16 + (lane>>4)*8 + (j&7)][w*16 + (lane&15)]
// so the GEMM kernel loads a v16bf per lane with one contiguous 32B read.
// ---------------------------------------------------------------------------
__global__ __launch_bounds__(256) void pack_w_kernel(const float* __restrict__ W,
                                                     __bf16* __restrict__ Wp,
                                                     int ktiles) {
    int idx = blockIdx.x * blockDim.x + threadIdx.x;
    int total = ktiles * 8 * 32 * 16;
    if (idx >= total) return;
    int j    = idx & 15;
    int lane = (idx >> 4) & 31;
    int w    = (idx >> 9) & 7;
    int kt   = idx >> 12;
    int k = kt * 32 + (j >> 3) * 16 + (lane >> 4) * 8 + (j & 7);
    int n = w * 16 + (lane & 15);
    Wp[idx] = (__bf16)W[k * F + n];
}

// ---------------------------------------------------------------------------
// Fused gather + MLP(512->128, LeakyReLU 0.1, 128->128) over 32 edges / block.
// 8 wave32 per block; wave w owns output columns [16w, 16w+16).
// ---------------------------------------------------------------------------
__global__ __launch_bounds__(256) void edge_mlp_kernel(
    const float* __restrict__ x_s, const float* __restrict__ x_t,
    const long long* __restrict__ edge_index, const float* __restrict__ edge_attr,
    const float* __restrict__ u, const long long* __restrict__ batch_e,
    const __bf16* __restrict__ W1p, const float* __restrict__ b1,
    const __bf16* __restrict__ W2p, const float* __restrict__ b2,
    float* __restrict__ out, int E)
{
    __shared__ __bf16 hbuf[ROWS * HPITCH];    // gathered+concat features, bf16
    __shared__ __bf16 h2buf[ROWS * H2PITCH];  // post-activation intermediate
    __shared__ int sIdx[ROWS], tIdx[ROWS], bIdx[ROWS];

    const int tid    = threadIdx.x;
    const int lane   = tid & 31;
    const int w      = tid >> 5;
    const int laneLo = lane & 15;
    const int laneHi = lane >> 4;
    const int e0     = blockIdx.x * ROWS;

    // ---- stage gather indices (int64 -> int) ----
    if (tid < ROWS) {
        int e = e0 + tid;
        if (e < E) {
            sIdx[tid] = (int)edge_index[e];
            tIdx[tid] = (int)edge_index[(size_t)E + e];
            bIdx[tid] = (int)batch_e[e];
        } else {
            sIdx[tid] = 0; tIdx[tid] = 0; bIdx[tid] = 0;
        }
    }
    __syncthreads();

    // ---- cooperative coalesced gather: 32 rows x 512 f32 -> bf16 LDS ----
    #pragma unroll
    for (int i = 0; i < (ROWS * (KTOT / 4)) / 256; ++i) {  // 16 iters
        int c    = tid + i * 256;
        int row  = c >> 7;       // 128 float4 per row
        int col4 = c & 127;
        int seg  = col4 >> 5;    // 0:x_s 1:x_t 2:edge_attr 3:u
        int off  = (col4 & 31) * 4;
        int e    = e0 + row;
        float4 v = make_float4(0.f, 0.f, 0.f, 0.f);
        if (e < E) {
            const float* p = x_s;
            if (seg == 0)      p = x_s + (size_t)sIdx[row] * F + off;
            else if (seg == 1) p = x_t + (size_t)tIdx[row] * F + off;
            else if (seg == 2) p = edge_attr + (size_t)e * F + off;
            else               p = u + (size_t)bIdx[row] * F + off;
            v = *(const float4*)p;
        }
        bf16x4 pk;
        pk.x = (__bf16)v.x; pk.y = (__bf16)v.y;
        pk.z = (__bf16)v.z; pk.w = (__bf16)v.w;
        *(bf16x4*)&hbuf[row * HPITCH + col4 * 4] = pk;
    }
    __syncthreads();

    // ---- GEMM1: [32 x 512] x [512 x 128] ----
    v8f c0 = {}; v8f c1 = {};
    for (int kt = 0; kt < KTOT / 32; ++kt) {   // 16 iters
        v16bf bfrag = *(const v16bf*)(W1p + (((kt * 8 + w) * 32 + lane) << 4));
        int koff = kt * 32 + laneHi * 8;
        union { v16bf v; bf16x8 h[2]; } a0, a1;
        const __bf16* r0 = &hbuf[laneLo * HPITCH + koff];
        const __bf16* r1 = &hbuf[(16 + laneLo) * HPITCH + koff];
        a0.h[0] = *(const bf16x8*)(r0);
        a0.h[1] = *(const bf16x8*)(r0 + 16);
        a1.h[0] = *(const bf16x8*)(r1);
        a1.h[1] = *(const bf16x8*)(r1 + 16);
        c0 = __builtin_amdgcn_wmma_f32_16x16x32_bf16(false, a0.v, false, bfrag,
                                                     (short)0, c0, false, false);
        c1 = __builtin_amdgcn_wmma_f32_16x16x32_bf16(false, a1.v, false, bfrag,
                                                     (short)0, c1, false, false);
    }

    // ---- bias + LeakyReLU(0.1) -> bf16 LDS (intermediate) ----
    const int colN = w * 16 + laneLo;
    const float b1v = b1[colN];
    #pragma unroll
    for (int i = 0; i < 8; ++i) {
        int rl = i + laneHi * 8;   // local row within 16-row tile
        float x0 = c0[i] + b1v; x0 = x0 > 0.f ? x0 : 0.1f * x0;
        h2buf[rl * H2PITCH + colN] = (__bf16)x0;
        float x1 = c1[i] + b1v; x1 = x1 > 0.f ? x1 : 0.1f * x1;
        h2buf[(16 + rl) * H2PITCH + colN] = (__bf16)x1;
    }
    __syncthreads();

    // ---- GEMM2: [32 x 128] x [128 x 128] ----
    v8f d0 = {}; v8f d1 = {};
    #pragma unroll
    for (int kt = 0; kt < F / 32; ++kt) {      // 4 iters
        v16bf bfrag = *(const v16bf*)(W2p + (((kt * 8 + w) * 32 + lane) << 4));
        int koff = kt * 32 + laneHi * 8;
        union { v16bf v; bf16x8 h[2]; } a0, a1;
        const __bf16* r0 = &h2buf[laneLo * H2PITCH + koff];
        const __bf16* r1 = &h2buf[(16 + laneLo) * H2PITCH + koff];
        a0.h[0] = *(const bf16x8*)(r0);
        a0.h[1] = *(const bf16x8*)(r0 + 16);
        a1.h[0] = *(const bf16x8*)(r1);
        a1.h[1] = *(const bf16x8*)(r1 + 16);
        d0 = __builtin_amdgcn_wmma_f32_16x16x32_bf16(false, a0.v, false, bfrag,
                                                     (short)0, d0, false, false);
        d1 = __builtin_amdgcn_wmma_f32_16x16x32_bf16(false, a1.v, false, bfrag,
                                                     (short)0, d1, false, false);
    }

    // ---- bias + store fp32 output ----
    const float b2v = b2[colN];
    #pragma unroll
    for (int i = 0; i < 8; ++i) {
        int rl = i + laneHi * 8;
        int ea = e0 + rl;
        if (ea < E) out[(size_t)ea * F + colN] = d0[i] + b2v;
        int eb = e0 + 16 + rl;
        if (eb < E) out[(size_t)eb * F + colN] = d1[i] + b2v;
    }
}

// ---------------------------------------------------------------------------
extern "C" void kernel_launch(void* const* d_in, const int* in_sizes, int n_in,
                              void* d_out, int out_size, void* d_ws, size_t ws_size,
                              hipStream_t stream) {
    const float*     x_s        = (const float*)d_in[0];
    const float*     x_t        = (const float*)d_in[1];
    const long long* edge_index = (const long long*)d_in[2];
    const float*     edge_attr  = (const float*)d_in[3];
    const float*     u          = (const float*)d_in[4];
    const long long* batch_e    = (const long long*)d_in[5];
    const float*     W1         = (const float*)d_in[6];
    const float*     b1         = (const float*)d_in[7];
    const float*     W2         = (const float*)d_in[8];
    const float*     b2         = (const float*)d_in[9];
    float*           out        = (float*)d_out;

    const int E = in_sizes[5];   // batch_e has E elements

    // workspace: packed bf16 weights (W1p: 128 KB, W2p: 32 KB)
    __bf16* W1p = (__bf16*)d_ws;
    __bf16* W2p = (__bf16*)((char*)d_ws + (size_t)KTOT * F * sizeof(__bf16));

    pack_w_kernel<<<(KTOT * F + 255) / 256, 256, 0, stream>>>(W1, W1p, KTOT / 32);
    pack_w_kernel<<<(F * F + 255) / 256, 256, 0, stream>>>(W2, W2p, F / 32);

    int nblk = (E + ROWS - 1) / ROWS;   // 15625 for E = 500000
    edge_mlp_kernel<<<nblk, 256, 0, stream>>>(x_s, x_t, edge_index, edge_attr, u,
                                              batch_e, W1p, b1, W2p, b2, out, E);
}
